// SelfAttention_46377056862597
// MI455X (gfx1250) — matrix operand order repaired and checked by reading, not compiled, not run on hardware
//
#include <hip/hip_runtime.h>
#include <hip/hip_bf16.h>
#include <math.h>

typedef __bf16 bf16_t;
typedef __bf16 v16bf __attribute__((ext_vector_type(16)));
typedef __bf16 v4bf  __attribute__((ext_vector_type(4)));
typedef float  v8f   __attribute__((ext_vector_type(8)));
typedef unsigned int v4u __attribute__((ext_vector_type(4)));
typedef int v8i __attribute__((ext_vector_type(8)));
typedef int v4i __attribute__((ext_vector_type(4)));

#define WMMA_BF16(a, b, c) \
  __builtin_amdgcn_wmma_f32_16x16x32_bf16(false, (a), false, (b), (short)0, (c), false, false)

union FragU { uint4 u[2]; v16bf v; };

static __device__ __forceinline__ v16bf load_frag16(const bf16_t* p0, const bf16_t* p1) {
  FragU f;
  f.u[0] = *(const uint4*)p0;
  f.u[1] = *(const uint4*)p1;
  return f.v;
}

static __device__ __forceinline__ v8f zero_v8f() {
  v8f z;
#pragma unroll
  for (int i = 0; i < 8; ++i) z[i] = 0.0f;
  return z;
}

// ---------------------------------------------------------------------------
// TDM: DMA a 2D bf16 tile [rows x rowlen] (row stride = strideElems) from
// global memory into LDS at byte offset lds_off, inserting 16B of padding
// after every 64B row (=> LDS row stride of 40 bf16 = 80B).
// Descriptor layout per CDNA5 ISA.
// ---------------------------------------------------------------------------
static __device__ __forceinline__ void tdm_load_tile_2d(
    unsigned lds_off, const bf16_t* gptr, int rows, int rowlen,
    int strideElems, int tdim0, int tdim1) {
  unsigned long long ga = (unsigned long long)(uintptr_t)gptr;
  v4u g0;
  g0[0] = 1u;                                   // count=1, user mode
  g0[1] = lds_off;                              // lds_addr (bytes)
  g0[2] = (unsigned)(ga & 0xFFFFFFFFu);         // global_addr[31:0]
  g0[3] = (unsigned)((ga >> 32) & 0x1FFFFFFu)   // global_addr[56:32]
          | (2u << 30);                         // type=2 ("image")
  v8i g1;
  g1[0] = (int)((1u << 16)      // data_size=1 -> 2 bytes
                | (1u << 20)    // pad_enable
                | (3u << 22)    // pad_interval: 16 DWORDs = 64B
                | (3u << 25));  // pad_amount: 4 DWORDs = 16B
  g1[1] = (int)(((unsigned)tdim0 & 0xFFFFu) << 16);                 // tensor_dim0 lo16
  g1[2] = (int)(((unsigned)tdim0 >> 16) |
                (((unsigned)tdim1 & 0xFFFFu) << 16));               // dim0 hi / dim1 lo
  g1[3] = (int)(((unsigned)tdim1 >> 16) |
                ((unsigned)rowlen << 16));                          // dim1 hi / tile_dim0
  g1[4] = rows;          // tile_dim1 (tile_dim2 = 0 -> 2D)
  g1[5] = strideElems;   // tensor_dim0_stride[31:0]
  g1[6] = 0;             // stride hi | dim1_stride lo (unused for 2D)
  g1[7] = 0;
  v4i z4; z4[0] = z4[1] = z4[2] = z4[3] = 0;
#if defined(__clang_major__) && (__clang_major__ >= 23)
  v8i z8;
#pragma unroll
  for (int i = 0; i < 8; ++i) z8[i] = 0;
  __builtin_amdgcn_tensor_load_to_lds(g0, g1, z4, z4, z8, 0);
#else
  __builtin_amdgcn_tensor_load_to_lds(g0, g1, z4, z4, 0);
#endif
}

static __device__ __forceinline__ unsigned lds_offset_of(const void* p) {
  // Generic LDS addresses carry the LDS byte offset in addr[31:0].
  return (unsigned)(unsigned long long)(uintptr_t)p;
}

// ---------------------------------------------------------------------------
// f32 -> bf16 conversion (vectorized 4-wide)
// ---------------------------------------------------------------------------
__global__ void cvt_f32_to_bf16_k(const float* __restrict__ src,
                                  bf16_t* __restrict__ dst, int n4) {
  int i = blockIdx.x * blockDim.x + threadIdx.x;
  if (i < n4) {
    float4 v = ((const float4*)src)[i];
    v4bf t;
    t[0] = (bf16_t)v.x; t[1] = (bf16_t)v.y; t[2] = (bf16_t)v.z; t[3] = (bf16_t)v.w;
    ((v4bf*)dst)[i] = t;
  }
}

// f32 [K,N] -> bf16 transposed [N,K] (one-time weight transform so the GEMM's
// B tiles are plain row-major 2D TDM tiles).
__global__ void cvt_transpose_bf16_k(const float* __restrict__ src,
                                     bf16_t* __restrict__ dst, int K, int N) {
  int idx = blockIdx.x * blockDim.x + threadIdx.x;
  if (idx < K * N) {
    int k = idx / N, n = idx % N;
    dst[(size_t)n * K + k] = (bf16_t)src[idx];
  }
}

// ---------------------------------------------------------------------------
// Tiled bf16 WMMA GEMM with TDM double-buffered LDS staging.
// C[M,N] = A[M,K] * BT[N,K]^T + bias
// MODE 0: QKV epilogue -> scatter bf16 Q[B,H,T,D], K[B,H,T,D], Vt[B,H,D,T]
// MODE 1: f32 output   -> outF[M,N]
// Block: 256 threads (8 waves), tile 128x128, K-step 32.
// ---------------------------------------------------------------------------
template <int MODE>
__global__ __launch_bounds__(256) void gemm_bf16_wmma_k(
    const bf16_t* __restrict__ A, const bf16_t* __restrict__ BT,
    const float* __restrict__ bias, float* __restrict__ outF,
    bf16_t* __restrict__ Qo, bf16_t* __restrict__ Ko, bf16_t* __restrict__ Vto,
    int M, int N, int K) {
  constexpr int BM = 128, BN = 128, BK = 32, LDAs = BK + 8;  // 40 bf16 = 80B rows
  __shared__ bf16_t As[2][BM][LDAs];    // [buf][row][k]
  __shared__ bf16_t BsT[2][BN][LDAs];   // [buf][col][k]

  const int tid = threadIdx.x;
  const int lane = tid & 31;
  const int wave = tid >> 5;
  const int col = lane & 15;
  const int grp = lane >> 4;        // 0 or 1
  const int waveM = wave >> 2;      // 0..1 -> 64-row slab
  const int waveN = wave & 3;       // 0..3 -> 32-col slab
  const int rowBase = blockIdx.y * BM;
  const int colBase = blockIdx.x * BN;
  const int nk = K / BK;

  v8f acc[4][2];
#pragma unroll
  for (int mi = 0; mi < 4; ++mi)
#pragma unroll
    for (int nj = 0; nj < 2; ++nj) acc[mi][nj] = zero_v8f();

  auto issue_tile = [&](int i) {
    const int kk = i * BK;
    const int buf = i & 1;
    tdm_load_tile_2d(lds_offset_of(&As[buf][0][0]),
                     A + (size_t)rowBase * K + kk, BM, BK, K, K, M);
    tdm_load_tile_2d(lds_offset_of(&BsT[buf][0][0]),
                     BT + (size_t)colBase * K + kk, BN, BK, K, K, N);
  };

  if (tid < 32) issue_tile(0);

  for (int i = 0; i < nk; ++i) {
    const int buf = i & 1;
    if (tid < 32) {
      if (i + 1 < nk) {
        issue_tile(i + 1);                          // overlap next DMA w/ compute
        __builtin_amdgcn_s_wait_tensorcnt(2);       // tile i complete (in-order)
      } else {
        __builtin_amdgcn_s_wait_tensorcnt(0);
      }
    }
    __syncthreads();   // tile i visible to all waves

    v16bf af[4], bfr[2];
#pragma unroll
    for (int mi = 0; mi < 4; ++mi) {
      const bf16_t* p = &As[buf][waveM * 64 + mi * 16 + col][0];
      af[mi] = load_frag16(p + grp * 16, p + grp * 16 + 8);
    }
#pragma unroll
    for (int nj = 0; nj < 2; ++nj) {
      const bf16_t* p = &BsT[buf][waveN * 32 + nj * 16 + col][grp * 16];
      bfr[nj] = load_frag16(p, p + 8);
    }
#pragma unroll
    for (int mi = 0; mi < 4; ++mi)
#pragma unroll
      for (int nj = 0; nj < 2; ++nj)
        acc[mi][nj] = WMMA_BF16(af[mi], bfr[nj], acc[mi][nj]);

    __syncthreads();   // all done reading buf before wave0 overwrites it (i+2)
  }

  // Epilogue
#pragma unroll
  for (int mi = 0; mi < 4; ++mi) {
#pragma unroll
    for (int nj = 0; nj < 2; ++nj) {
      const int gcol = colBase + waveN * 32 + nj * 16 + col;
      const float bv = bias[gcol];
      if (MODE == 0) {
        const int which = gcol >> 10;      // 0=q, 1=k, 2=v
        const int e = gcol & 1023;
        const int h = e >> 6, d = e & 63;
#pragma unroll
        for (int r = 0; r < 8; ++r) {
          const int grow = rowBase + waveM * 64 + mi * 16 + r + 8 * grp;
          const int b = grow >> 11, t = grow & 2047;
          const float val = acc[mi][nj][r] + bv;
          const size_t bh = (size_t)(b * 16 + h);
          if (which == 0)      Qo[(bh * 2048 + t) * 64 + d] = (bf16_t)val;
          else if (which == 1) Ko[(bh * 2048 + t) * 64 + d] = (bf16_t)val;
          else                 Vto[(bh * 64 + d) * 2048 + t] = (bf16_t)val;
        }
      } else {
#pragma unroll
        for (int r = 0; r < 8; ++r) {
          const int grow = rowBase + waveM * 64 + mi * 16 + r + 8 * grp;
          outF[(size_t)grow * N + gcol] = acc[mi][nj][r] + bv;
        }
      }
    }
  }
}

// ---------------------------------------------------------------------------
// Flash attention: 1 wave per 16-query tile, 32-key chunks, causal.
// Q,K: [B*H, T, 64] bf16; Vt: [B*H, 64, T] bf16; out: [B*T, 1024] bf16.
// Block: 128 threads = 4 independent waves (no in-loop barriers).
// ---------------------------------------------------------------------------
__global__ __launch_bounds__(128) void flash_attn_wmma_k(
    const bf16_t* __restrict__ Q, const bf16_t* __restrict__ Kt,
    const bf16_t* __restrict__ Vt, bf16_t* __restrict__ attnOut) {
  constexpr int T = 2048, D = 64;
  __shared__ bf16_t Ps[4][16][40];  // per-wave P bounce buffer (16x32 +pad)

  const int lane = threadIdx.x & 31;
  const int wv = threadIdx.x >> 5;
  const int col = lane & 15, grp = lane >> 4;
  const int bh = blockIdx.y;
  const int b = bh >> 4, h = bh & 15;
  const int qbase = (blockIdx.x * 4 + wv) * 16;

  const bf16_t* Qh = Q + (size_t)bh * T * D;
  const bf16_t* Kh = Kt + (size_t)bh * T * D;
  const bf16_t* Vh = Vt + (size_t)bh * D * T;

  // Q A-fragments for the two K=32 depth steps (d 0..31, 32..63)
  const bf16_t* qp = Qh + (size_t)(qbase + col) * D;
  const v16bf aq0 = load_frag16(qp + grp * 16, qp + grp * 16 + 8);
  const v16bf aq1 = load_frag16(qp + 32 + grp * 16, qp + 32 + grp * 16 + 8);

  float m[8], l[8];
  v8f o[4];
#pragma unroll
  for (int r = 0; r < 8; ++r) { m[r] = -INFINITY; l[r] = 0.0f; }
#pragma unroll
  for (int j = 0; j < 4; ++j) o[j] = zero_v8f();

  const float scale = 0.125f;  // 1/sqrt(64)
  const int kcEnd = (qbase + 15) >> 5;
  for (int kc = 0; kc <= kcEnd; ++kc) {
    const int keyBase = kc << 5;
    if (kc < kcEnd) {  // prefetch next K/V chunk (global_prefetch_b8)
      __builtin_prefetch(Kh + (size_t)(keyBase + 32) * D, 0, 1);
      __builtin_prefetch(Vh + keyBase + 32, 0, 1);
    }

    // K^T B-fragments: bk[key-subtile][depth-half]
    v16bf bk[2][2];
#pragma unroll
    for (int n = 0; n < 2; ++n) {
      const bf16_t* kp = Kh + (size_t)(keyBase + n * 16 + col) * D;
      bk[n][0] = load_frag16(kp + grp * 16, kp + grp * 16 + 8);
      bk[n][1] = load_frag16(kp + 32 + grp * 16, kp + 32 + grp * 16 + 8);
    }

    v8f s0 = WMMA_BF16(aq0, bk[0][0], zero_v8f());
    s0 = WMMA_BF16(aq1, bk[0][1], s0);
    v8f s1 = WMMA_BF16(aq0, bk[1][0], zero_v8f());
    s1 = WMMA_BF16(aq1, bk[1][1], s1);

    // Online softmax. C-tile row = r + 8*grp (shared by the 16 lanes of a half).
    float p0[8], p1[8];
#pragma unroll
    for (int r = 0; r < 8; ++r) {
      const int qr = qbase + r + 8 * grp;
      float v0 = s0[r] * scale, v1 = s1[r] * scale;
      if (keyBase + col > qr) v0 = -INFINITY;
      if (keyBase + 16 + col > qr) v1 = -INFINITY;
      float mx = fmaxf(v0, v1);
      mx = fmaxf(mx, __shfl_xor(mx, 1, 32));
      mx = fmaxf(mx, __shfl_xor(mx, 2, 32));
      mx = fmaxf(mx, __shfl_xor(mx, 4, 32));
      mx = fmaxf(mx, __shfl_xor(mx, 8, 32));
      const float mn = fmaxf(m[r], mx);
      const float alpha = __expf(m[r] - mn);
      const float e0 = __expf(v0 - mn);
      const float e1 = __expf(v1 - mn);
      float rs = e0 + e1;
      rs += __shfl_xor(rs, 1, 32);
      rs += __shfl_xor(rs, 2, 32);
      rs += __shfl_xor(rs, 4, 32);
      rs += __shfl_xor(rs, 8, 32);
      l[r] = l[r] * alpha + rs;
      m[r] = mn;
      p0[r] = e0; p1[r] = e1;
#pragma unroll
      for (int j = 0; j < 4; ++j) o[j][r] *= alpha;
    }

    // C-layout -> A-layout for P via per-wave LDS bounce
#pragma unroll
    for (int r = 0; r < 8; ++r) {
      const int lr = r + 8 * grp;
      Ps[wv][lr][col] = (bf16_t)p0[r];
      Ps[wv][lr][16 + col] = (bf16_t)p1[r];
    }
    asm volatile("s_wait_dscnt 0" ::: "memory");
    const v16bf pa =
        load_frag16(&Ps[wv][col][grp * 16], &Ps[wv][col][grp * 16 + 8]);

    // O += P * V  (V transposed in memory -> contiguous B-fragments)
#pragma unroll
    for (int j = 0; j < 4; ++j) {
      const bf16_t* vp = Vh + (size_t)(j * 16 + col) * T + keyBase + grp * 16;
      const v16bf bv = load_frag16(vp, vp + 8);
      o[j] = WMMA_BF16(pa, bv, o[j]);
    }
  }

  // Normalize and write [B*T, E] bf16 (col = h*64 + d)
#pragma unroll
  for (int r = 0; r < 8; ++r) {
    const int row = qbase + r + 8 * grp;
    const float inv = 1.0f / l[r];
    const size_t base = ((size_t)b * T + row) * 1024 + h * 64;
#pragma unroll
    for (int j = 0; j < 4; ++j)
      attnOut[base + j * 16 + col] = (bf16_t)(o[j][r] * inv);
  }
}

// ---------------------------------------------------------------------------
extern "C" void kernel_launch(void* const* d_in, const int* in_sizes, int n_in,
                              void* d_out, int out_size, void* d_ws,
                              size_t ws_size, hipStream_t stream) {
  const float* x = (const float*)d_in[0];       // [2,2048,1024]
  const float* W_attn = (const float*)d_in[1];  // [1024,3072]
  const float* b_attn = (const float*)d_in[2];  // [3072]
  const float* W_proj = (const float*)d_in[3];  // [1024,1024]
  const float* b_proj = (const float*)d_in[4];  // [1024]
  float* out = (float*)d_out;                   // [2,2048,1024]

  const int Mrows = 4096, E = 1024, N3 = 3072;
  char* ws = (char*)d_ws;
  const size_t MiB = 1024 * 1024;
  bf16_t* xbf   = (bf16_t*)(ws + 0);          //  8 MiB  [4096,1024]
  bf16_t* waT   = (bf16_t*)(ws + 8 * MiB);    //  6 MiB  [3072,1024] (W_attn^T)
  bf16_t* wpT   = (bf16_t*)(ws + 14 * MiB);   //  2 MiB  [1024,1024] (W_proj^T)
  bf16_t* Qbf   = (bf16_t*)(ws + 16 * MiB);   //  8 MiB  [32,2048,64]
  bf16_t* Kbf   = (bf16_t*)(ws + 24 * MiB);   //  8 MiB  [32,2048,64]
  bf16_t* Vtbf  = (bf16_t*)(ws + 32 * MiB);   //  8 MiB  [32,64,2048]
  bf16_t* attnb = (bf16_t*)(ws + 40 * MiB);   //  8 MiB  [4096,1024]

  // 1) f32 -> bf16 conversions (weights transposed once for TDM-friendly tiles)
  cvt_f32_to_bf16_k<<<(Mrows * E / 4 + 255) / 256, 256, 0, stream>>>(x, xbf, Mrows * E / 4);
  cvt_transpose_bf16_k<<<(E * N3 + 255) / 256, 256, 0, stream>>>(W_attn, waT, E, N3);
  cvt_transpose_bf16_k<<<(E * E + 255) / 256, 256, 0, stream>>>(W_proj, wpT, E, E);

  // 2) QKV GEMM: [4096,3072] = xbf @ waT^T + b_attn -> Q/K/Vt (bf16)
  gemm_bf16_wmma_k<0><<<dim3(N3 / 128, Mrows / 128), 256, 0, stream>>>(
      xbf, waT, b_attn, nullptr, Qbf, Kbf, Vtbf, Mrows, N3, E);

  // 3) causal flash attention -> attnb [4096,1024] bf16
  flash_attn_wmma_k<<<dim3(2048 / 64, 32), 128, 0, stream>>>(Qbf, Kbf, Vtbf, attnb);

  // 4) projection: out = attnb @ wpT^T + b_proj (f32)
  gemm_bf16_wmma_k<1><<<dim3(E / 128, Mrows / 128), 256, 0, stream>>>(
      attnb, wpT, b_proj, out, nullptr, nullptr, nullptr, Mrows, E, E);
}
